// GATBaseline_84232898609754
// MI455X (gfx1250) — matrix operand order repaired and checked by reading, not compiled, not run on hardware
//
#include <hip/hip_runtime.h>
#include <hip/hip_bf16.h>
#include <stdint.h>

// ---------------- problem constants (from reference) ----------------
#define NNODES 20000
#define NEDGES 320000
#define ETOT   (NEDGES + NNODES)   // edges + self loops = 340000
#define F_IN   5
#define HIDDEN 768
#define NHEAD  8
#define CHEAD  96
#define NLAYER 16
#define EMB    512
#define NGRAPH 16
#define NEG_SLOPE 0.2f

// ---------------- CDNA5 WMMA types ----------------
typedef __bf16 bf16_t;
typedef __attribute__((ext_vector_type(16))) __bf16  v16bf;
typedef __attribute__((ext_vector_type(8)))  float   v8f;
typedef __attribute__((ext_vector_type(4)))  unsigned int v4u;

union FragBF16 {       // 16 bf16 = 32 bytes = two 128-bit loads
    v16bf v;
    v4u   q[2];
};

// fast f32 atomic add -> global_atomic_add_f32 (no CAS loop)
__device__ __forceinline__ void atomAddF(float* p, float v) {
    unsafeAtomicAdd(p, v);
}

// =====================================================================
// 1) Projection: h = x @ proj_W + proj_b ; also emit bf16 copy of h
// =====================================================================
__global__ void gat_proj_kernel(const float* __restrict__ x,
                                const float* __restrict__ W,
                                const float* __restrict__ b,
                                float* __restrict__ h,
                                bf16_t* __restrict__ hb)
{
    int idx = blockIdx.x * blockDim.x + threadIdx.x;   // n*HIDDEN + c
    if (idx >= NNODES * HIDDEN) return;
    int n = idx / HIDDEN;
    int c = idx - n * HIDDEN;
    float s = b[c];
    #pragma unroll
    for (int f = 0; f < F_IN; ++f)
        s += x[n * F_IN + f] * W[f * HIDDEN + c];
    h[idx]  = s;
    hb[idx] = (bf16_t)s;
}

// =====================================================================
// 2) Weight convert + transpose: Wt[n][k] = bf16(W[k][n])  (768x768)
// =====================================================================
__global__ void gat_wcvt_kernel(const float* __restrict__ W,
                                bf16_t* __restrict__ Wt)
{
    int idx = blockIdx.x * blockDim.x + threadIdx.x;   // k*HIDDEN + n
    if (idx >= HIDDEN * HIDDEN) return;
    int k = idx / HIDDEN;
    int n = idx - k * HIDDEN;
    Wt[(size_t)n * HIDDEN + k] = (bf16_t)W[idx];
}

// =====================================================================
// 3) WMMA GEMM with LDS-staged B panel.
//    C[M,768] = A[M,768](bf16) @ B ; B supplied transposed Bt[n][k].
//    Block = 8 waves; wave w -> rows [by*128 + w*16, +16), cols [bx*64, +64).
//    B panel (64 cols x 768 K) staged to LDS in 2 chunks of K=384 (49 KB),
//    each loaded once per block instead of once per wave (8x traffic cut).
// =====================================================================
#define KCHUNK   384
#define BN       64
#define LDS_PAD  8                       // 8 bf16 = 16 B row pad (bank stagger)
#define LDS_STR  (KCHUNK + LDS_PAD)      // 392 bf16 per row

__global__ __launch_bounds__(256)
void gat_gemm_wmma_kernel(const bf16_t* __restrict__ A,
                          const bf16_t* __restrict__ Bt,
                          float* __restrict__ C,
                          int M)
{
    __shared__ bf16_t Bs[BN * LDS_STR];  // 64*392*2 = 50176 bytes

    const int tid   = threadIdx.x;
    const int lane  = tid & 31;
    const int wave  = tid >> 5;
    const int mtile = blockIdx.y * 8 + wave;   // 16-row tile index (may overrun)
    const int ncol0 = blockIdx.x * BN;         // first of 64 output columns
    const int nlo   = lane & 15;
    const int khalf = lane >> 4;               // 0/1: ISA half-wave K split

    // clamp A row so overrun tiles still compute with full EXEC (stores guarded)
    int mrow = mtile * 16 + nlo;
    if (mrow >= M) mrow = M - 1;
    const bf16_t* arow = A + (size_t)mrow * HIDDEN;

    v8f acc[4];
    #pragma unroll
    for (int j = 0; j < 4; ++j)
        #pragma unroll
        for (int r = 0; r < 8; ++r) acc[j][r] = 0.0f;

    for (int kc = 0; kc < HIDDEN / KCHUNK; ++kc) {
        const int kbase = kc * KCHUNK;
        if (kc) __syncthreads();               // protect previous chunk reads

        // cooperative fill: 64 rows x 48 chunks of 16B = 3072 v4u, 12 per thread
        for (int i = tid; i < BN * (KCHUNK / 8); i += 256) {
            int r  = i / (KCHUNK / 8);
            int c8 = i - r * (KCHUNK / 8);
            *(v4u*)(&Bs[r * LDS_STR + c8 * 8]) =
                *(const v4u*)(Bt + (size_t)(ncol0 + r) * HIDDEN + kbase + c8 * 8);
        }
        __syncthreads();

        for (int k2 = 0; k2 < KCHUNK; k2 += 32) {
            // 16-bit A 16x32 layout: lanes 0-15 K {k..k+7, k+16..k+23},
            // lanes 16-31 K {k+8..k+15, k+24..k+31}
            FragBF16 fa;
            fa.q[0] = *(const v4u*)(arow + kbase + k2 +      khalf * 8);
            fa.q[1] = *(const v4u*)(arow + kbase + k2 + 16 + khalf * 8);

            #pragma unroll
            for (int j = 0; j < 4; ++j) {
                // 16-bit B 32x16: lane -> col N, lanes 0-15 K {k..k+7, k+16..k+23},
                // lanes 16-31 K {k+8..k+15, k+24..k+31} (matches A)
                const bf16_t* bs = &Bs[(j * 16 + nlo) * LDS_STR + k2 + khalf * 8];
                FragBF16 fb;
                fb.q[0] = *(const v4u*)(bs);
                fb.q[1] = *(const v4u*)(bs + 16);
                acc[j] = __builtin_amdgcn_wmma_f32_16x16x32_bf16(
                             false, fa.v, false, fb.v,
                             (short)0, acc[j], false, false);
            }
        }
    }

    // C/D layout: VGPR r -> M = mtile*16 + (lane>>4)*8 + r, N = lane&15
    const int mbase = mtile * 16 + khalf * 8;
    #pragma unroll
    for (int j = 0; j < 4; ++j) {
        const int ncol = ncol0 + j * 16 + nlo;
        #pragma unroll
        for (int r = 0; r < 8; ++r) {
            int row = mbase + r;
            if (row < M)
                C[(size_t)row * HIDDEN + ncol] = acc[j][r];
        }
    }
}

// =====================================================================
// 4) Attention dots per (node, head) + init max/denom
// =====================================================================
__global__ void gat_att_kernel(const float* __restrict__ g,
                               const float* __restrict__ att_s,
                               const float* __restrict__ att_d,
                               float* __restrict__ a_s,
                               float* __restrict__ a_d,
                               int*   __restrict__ m_int,
                               float* __restrict__ denom)
{
    int idx = blockIdx.x * blockDim.x + threadIdx.x;   // n*NHEAD + h
    if (idx >= NNODES * NHEAD) return;
    int n  = idx >> 3;
    int hh = idx & 7;
    const float* grow = g + (size_t)n * HIDDEN + hh * CHEAD;
    const float* vs   = att_s + hh * CHEAD;
    const float* vd   = att_d + hh * CHEAD;
    float ss = 0.f, sd = 0.f;
    #pragma unroll 8
    for (int c = 0; c < CHEAD; ++c) {
        float v = grow[c];
        ss += v * vs[c];
        sd += v * vd[c];
    }
    a_s[idx] = ss;
    a_d[idx] = sd;
    m_int[idx] = (int)0x80000000;   // minimum ordered-float key
    denom[idx] = 0.f;
}

// =====================================================================
// 5) Zero the aggregation accumulator
// =====================================================================
__global__ void gat_zero_kernel(float* __restrict__ p, int count)
{
    int idx = blockIdx.x * blockDim.x + threadIdx.x;
    if (idx < count) p[idx] = 0.f;
}

// ---- ordered-float <-> int monotonic mapping (involution) ----
__device__ __forceinline__ int f32_to_ordkey(float f) {
    int b = __float_as_int(f);
    return b >= 0 ? b : (b ^ 0x7fffffff);
}
__device__ __forceinline__ float ordkey_to_f32(int k) {
    int b = k >= 0 ? k : (k ^ 0x7fffffff);
    return __int_as_float(b);
}

// =====================================================================
// 6) Edge pass 1: e = leaky_relu(a_s[src]+a_d[dst]); segment max via
//    atomicMax on monotonically-mapped int key
// =====================================================================
__global__ void gat_edge_max_kernel(const int* __restrict__ ei,
                                    const float* __restrict__ a_s,
                                    const float* __restrict__ a_d,
                                    float* __restrict__ ebuf,
                                    int*   __restrict__ m_int)
{
    int idx = blockIdx.x * blockDim.x + threadIdx.x;   // e*NHEAD + h
    if (idx >= ETOT * NHEAD) return;
    int e  = idx >> 3;
    int hh = idx & 7;
    int s, d;
    if (e < NEDGES) { s = ei[e]; d = ei[NEDGES + e]; }
    else            { s = d = e - NEDGES; }
    float v = a_s[s * NHEAD + hh] + a_d[d * NHEAD + hh];
    v = v > 0.f ? v : NEG_SLOPE * v;
    ebuf[idx] = v;
    atomicMax(&m_int[d * NHEAD + hh], f32_to_ordkey(v));
}

// =====================================================================
// 7) Edge pass 2: p = exp(e - m[dst]); denom[dst] += p
// =====================================================================
__global__ void gat_edge_exp_kernel(const int* __restrict__ ei,
                                    const float* __restrict__ ebuf,
                                    const int* __restrict__ m_int,
                                    float* __restrict__ pbuf,
                                    float* __restrict__ denom)
{
    int idx = blockIdx.x * blockDim.x + threadIdx.x;   // e*NHEAD + h
    if (idx >= ETOT * NHEAD) return;
    int e  = idx >> 3;
    int hh = idx & 7;
    int d = (e < NEDGES) ? ei[NEDGES + e] : (e - NEDGES);
    float m = ordkey_to_f32(m_int[d * NHEAD + hh]);
    float p = __expf(ebuf[idx] - m);
    pbuf[idx] = p;
    atomAddF(&denom[d * NHEAD + hh], p);
}

// =====================================================================
// 8) Aggregation: out[dst] += alpha * g[src]  (one wave32 per edge)
// =====================================================================
__global__ __launch_bounds__(256)
void gat_aggregate_kernel(const int* __restrict__ ei,
                          const float* __restrict__ g,
                          const float* __restrict__ pbuf,
                          const float* __restrict__ denom,
                          float* __restrict__ out)
{
    int gid  = blockIdx.x * blockDim.x + threadIdx.x;
    int e    = gid >> 5;            // one wave per edge
    int lane = threadIdx.x & 31;
    if (e >= ETOT) return;
    int s, d;
    if (e < NEDGES) { s = ei[e]; d = ei[NEDGES + e]; }
    else            { s = d = e - NEDGES; }

    float alpha[NHEAD];
    #pragma unroll
    for (int hh = 0; hh < NHEAD; ++hh)
        alpha[hh] = pbuf[e * NHEAD + hh] / (denom[d * NHEAD + hh] + 1e-16f);

    const float* grow = g + (size_t)s * HIDDEN;
    float* orow       = out + (size_t)d * HIDDEN;
    #pragma unroll
    for (int i = 0; i < HIDDEN / 32; ++i) {
        int c  = i * 32 + lane;
        int hh = c / CHEAD;
        atomAddF(&orow[c], alpha[hh] * grow[c]);
    }
}

// =====================================================================
// 9) h = relu(out + bias); also refresh bf16 copy for next layer's GEMM
// =====================================================================
__global__ void gat_bias_relu_kernel(const float* __restrict__ acc,
                                     const float* __restrict__ b,
                                     float* __restrict__ h,
                                     bf16_t* __restrict__ hb)
{
    int idx = blockIdx.x * blockDim.x + threadIdx.x;
    if (idx >= NNODES * HIDDEN) return;
    int c = idx % HIDDEN;
    float v = acc[idx] + b[c];
    v = v > 0.f ? v : 0.f;
    h[idx]  = v;
    hb[idx] = (bf16_t)v;
}

// =====================================================================
// 10) Pooling
// =====================================================================
__global__ void gat_pool_zero_kernel(float* __restrict__ pooled,
                                     float* __restrict__ cnt)
{
    int idx = blockIdx.x * blockDim.x + threadIdx.x;
    if (idx < NGRAPH * HIDDEN) pooled[idx] = 0.f;
    if (idx < NGRAPH) cnt[idx] = 0.f;
}

__global__ void gat_cnt_kernel(const int* __restrict__ batch,
                               float* __restrict__ cnt)
{
    int n = blockIdx.x * blockDim.x + threadIdx.x;
    if (n >= NNODES) return;
    atomAddF(&cnt[batch[n]], 1.f);
}

__global__ void gat_pool_sum_kernel(const float* __restrict__ h,
                                    const int* __restrict__ batch,
                                    float* __restrict__ pooled)
{
    int idx = blockIdx.x * blockDim.x + threadIdx.x;   // n*HIDDEN + c
    if (idx >= NNODES * HIDDEN) return;
    int n = idx / HIDDEN;
    int c = idx - n * HIDDEN;
    atomAddF(&pooled[batch[n] * HIDDEN + c], h[idx]);
}

// =====================================================================
// 11) Final FC: out[b,e] = (pooled[b]/max(cnt,1)) @ fc_W + fc_b
// =====================================================================
__global__ void gat_fc_kernel(const float* __restrict__ pooled,
                              const float* __restrict__ cnt,
                              const float* __restrict__ W,
                              const float* __restrict__ b,
                              float* __restrict__ outp)
{
    int idx = blockIdx.x * blockDim.x + threadIdx.x;   // b*EMB + e
    if (idx >= NGRAPH * EMB) return;
    int bb = idx / EMB;
    int e2 = idx - bb * EMB;
    float c = cnt[bb];
    float inv = 1.f / (c > 1.f ? c : 1.f);
    float s = b[e2];
    const float* prow = pooled + (size_t)bb * HIDDEN;
    for (int k = 0; k < HIDDEN; ++k)
        s += (prow[k] * inv) * W[(size_t)k * EMB + e2];
    outp[idx] = s;
}

// =====================================================================
// Host launcher
// =====================================================================
extern "C" void kernel_launch(void* const* d_in, const int* in_sizes, int n_in,
                              void* d_out, int out_size, void* d_ws, size_t ws_size,
                              hipStream_t stream)
{
    (void)in_sizes; (void)n_in; (void)out_size; (void)ws_size;

    const float* x        = (const float*)d_in[0];
    const int*   ei       = (const int*)  d_in[1];   // [2, E]
    const int*   batch    = (const int*)  d_in[2];
    const float* proj_W   = (const float*)d_in[3];
    const float* proj_b   = (const float*)d_in[4];
    const float* conv_W   = (const float*)d_in[5];   // [L, HID, HID]
    const float* att_src  = (const float*)d_in[6];   // [L, H, C]
    const float* att_dst  = (const float*)d_in[7];
    const float* conv_b   = (const float*)d_in[8];   // [L, HID]
    const float* fc_W     = (const float*)d_in[9];
    const float* fc_b     = (const float*)d_in[10];
    float* out = (float*)d_out;                      // [B, EMB]

    // ---- workspace layout (256B-aligned chunks) ----
    char* ws = (char*)d_ws;
    size_t off = 0;
    auto take = [&](size_t bytes) { char* p = ws + off; off += (bytes + 255) & ~size_t(255); return p; };

    float*  h_f32  = (float*) take((size_t)NNODES * HIDDEN * 4);
    bf16_t* h_bf16 = (bf16_t*)take((size_t)NNODES * HIDDEN * 2);
    float*  g_f32  = (float*) take((size_t)NNODES * HIDDEN * 4);
    float*  agg    = (float*) take((size_t)NNODES * HIDDEN * 4);
    bf16_t* Wt     = (bf16_t*)take((size_t)HIDDEN * HIDDEN * 2);
    float*  a_s    = (float*) take((size_t)NNODES * NHEAD * 4);
    float*  a_d    = (float*) take((size_t)NNODES * NHEAD * 4);
    int*    m_int  = (int*)   take((size_t)NNODES * NHEAD * 4);
    float*  denom  = (float*) take((size_t)NNODES * NHEAD * 4);
    float*  ebuf   = (float*) take((size_t)ETOT * NHEAD * 4);
    float*  pbuf   = (float*) take((size_t)ETOT * NHEAD * 4);
    float*  pooled = (float*) take((size_t)NGRAPH * HIDDEN * 4);
    float*  cnt    = (float*) take((size_t)NGRAPH * 4);

    const int TB = 256;
    const int nNH   = (NNODES * HIDDEN + TB - 1) / TB;     // 60000
    const int nWW   = (HIDDEN * HIDDEN + TB - 1) / TB;     // 2304
    const int nAtt  = (NNODES * NHEAD + TB - 1) / TB;
    const int nEdge = (ETOT * NHEAD + TB - 1) / TB;
    const int nAgg  = (ETOT * 32 + TB - 1) / TB;           // one wave32 per edge

    // GEMM grid: N=768 -> 12 blocks of 64 cols; M=20000 -> 1250 tiles / 8 waves
    dim3 ggrid(HIDDEN / BN, (NNODES / 16 + 7) / 8);

    // 1) projection
    gat_proj_kernel<<<nNH, TB, 0, stream>>>(x, proj_W, proj_b, h_f32, h_bf16);

    // 2) GAT layers
    for (int l = 0; l < NLAYER; ++l) {
        gat_wcvt_kernel<<<nWW, TB, 0, stream>>>(
            conv_W + (size_t)l * HIDDEN * HIDDEN, Wt);

        gat_gemm_wmma_kernel<<<ggrid, TB, 0, stream>>>(h_bf16, Wt, g_f32, NNODES);

        gat_att_kernel<<<nAtt, TB, 0, stream>>>(
            g_f32,
            att_src + (size_t)l * NHEAD * CHEAD,
            att_dst + (size_t)l * NHEAD * CHEAD,
            a_s, a_d, m_int, denom);

        gat_zero_kernel<<<nNH, TB, 0, stream>>>(agg, NNODES * HIDDEN);

        gat_edge_max_kernel<<<nEdge, TB, 0, stream>>>(ei, a_s, a_d, ebuf, m_int);

        gat_edge_exp_kernel<<<nEdge, TB, 0, stream>>>(ei, ebuf, m_int, pbuf, denom);

        gat_aggregate_kernel<<<nAgg, TB, 0, stream>>>(ei, g_f32, pbuf, denom, agg);

        gat_bias_relu_kernel<<<nNH, TB, 0, stream>>>(
            agg, conv_b + (size_t)l * HIDDEN, h_f32, h_bf16);
    }

    // 3) global mean pool + fc
    gat_pool_zero_kernel<<<(NGRAPH * HIDDEN + TB - 1) / TB, TB, 0, stream>>>(pooled, cnt);
    gat_cnt_kernel<<<(NNODES + TB - 1) / TB, TB, 0, stream>>>(batch, cnt);
    gat_pool_sum_kernel<<<nNH, TB, 0, stream>>>(h_f32, batch, pooled);
    gat_fc_kernel<<<(NGRAPH * EMB + TB - 1) / TB, TB, 0, stream>>>(pooled, cnt, fc_W, fc_b, out);
}
